// GraphTransformerPYG_4922032521873
// MI455X (gfx1250) — matrix operand order repaired and checked by reading, not compiled, not run on hardware
//
#include <hip/hip_runtime.h>
#include <hip/hip_bf16.h>
#include <math.h>

// ---------------------------------------------------------------------------
// Types for CDNA5 WMMA (wave32, 16x16x32 bf16 -> f32)
// ---------------------------------------------------------------------------
typedef __attribute__((ext_vector_type(16))) __bf16          v16bf;
typedef __attribute__((ext_vector_type(8)))  float           v8f;
typedef __attribute__((ext_vector_type(16))) unsigned short  v16us;
typedef __attribute__((ext_vector_type(8)))  unsigned short  v8us;

#define IN_CH   128
#define D1      256          // H*HID
#define DCAT    1024         // 4 * D1  (Q|K|V|S concatenated)
#define HEADS   4
#define HID_CH  64
#define OUT_CH  10
#define NGRAPH  64
#define NBLK    4            // N-tiles per wave (16x64 strip): A-fragment reuse

__device__ __forceinline__ unsigned short f2bf(float f) {
  unsigned u = __float_as_uint(f);
  unsigned r = (u >> 16) & 1u;
  u += 0x7FFFu + r;                 // round-to-nearest-even
  return (unsigned short)(u >> 16);
}
// order-preserving float<->uint encoding for atomic max
__device__ __forceinline__ unsigned encf(float f) {
  unsigned u = __float_as_uint(f);
  return (u & 0x80000000u) ? ~u : (u | 0x80000000u);
}
__device__ __forceinline__ float decf(unsigned u) {
  return (u & 0x80000000u) ? __uint_as_float(u & 0x7FFFFFFFu)
                           : __uint_as_float(~u);
}

// ---------------------------------------------------------------------------
// Weight prep: Wt[ocat*K + i] = (bf16) W_blk[i*256 + o];  bcat[ocat] = b_blk[o]
// ---------------------------------------------------------------------------
__global__ void prep_wt(const float* __restrict__ W0, const float* __restrict__ W1,
                        const float* __restrict__ W2, const float* __restrict__ W3,
                        const float* __restrict__ b0, const float* __restrict__ b1,
                        const float* __restrict__ b2, const float* __restrict__ b3,
                        unsigned short* __restrict__ Wt, float* __restrict__ bcat, int K) {
  int idx = blockIdx.x * blockDim.x + threadIdx.x;
  if (idx >= DCAT * K) return;
  int ocat = idx / K, i = idx - ocat * K;
  int blk = ocat >> 8, o = ocat & 255;
  const float* W = (blk == 0) ? W0 : (blk == 1) ? W1 : (blk == 2) ? W2 : W3;
  Wt[(size_t)ocat * K + i] = f2bf(W[(size_t)i * D1 + o]);
  if (i == 0) {
    const float* b = (blk == 0) ? b0 : (blk == 1) ? b1 : (blk == 2) ? b2 : b3;
    bcat[ocat] = b[o];
  }
}

__global__ void f32_to_bf16(const float* __restrict__ src,
                            unsigned short* __restrict__ dst, int n) {
  int i = blockIdx.x * blockDim.x + threadIdx.x;
  if (i < n) dst[i] = f2bf(src[i]);
}

// ---------------------------------------------------------------------------
// WMMA GEMM:  out[M x Nout] = A[M x K](bf16) @ Wt[Nout x K]^T(bf16) + bias
// One wave computes a 16x64 strip (4 x 16x16 tiles), reusing the A fragment
// 4x per K-step.  K stepped by 32 (v_wmma_f32_16x16x32_bf16).
// ---------------------------------------------------------------------------
__global__ void gemm_wmma_bf16(const unsigned short* __restrict__ A,
                               const unsigned short* __restrict__ Wt,
                               const float* __restrict__ bias,
                               float* __restrict__ out,
                               int M, int K, int Nout) {
  int wave = threadIdx.x >> 5;
  int lane = threadIdx.x & 31;
  int stripsN = Nout >> 6;                       // groups of 4 N-tiles
  int strip = blockIdx.x * (blockDim.x >> 5) + wave;
  int tilesM = (M + 15) >> 4;
  if (strip >= tilesM * stripsN) return;
  int sn = strip % stripsN, tm = strip / stripsN;
  int m0 = tm << 4, n0 = sn << 6;

  int lrow = lane & 15;
  int hi8  = (lane >> 4) << 3;   // 0 or 8
  int hi16 = hi8 << 1;           // 0 or 16

  // A fragment rows / B fragment columns; k = 16h + s per lane half (contiguous, same for A and B)
  const unsigned short* arow = A + (size_t)(m0 + lrow) * K + hi16;
  const unsigned short* brow[NBLK];
#pragma unroll
  for (int j = 0; j < NBLK; ++j)
    brow[j] = Wt + (size_t)(n0 + j * 16 + lrow) * K + hi16;

  v8f acc[NBLK];
#pragma unroll
  for (int j = 0; j < NBLK; ++j)
    acc[j] = (v8f){0.f, 0.f, 0.f, 0.f, 0.f, 0.f, 0.f, 0.f};

  for (int kk = 0; kk < K; kk += 32) {
    v8us alo = *(const v8us*)(arow + kk);        // K = hi16 + 0..7
    v8us ahi = *(const v8us*)(arow + kk + 8);    // K = hi16 + 8..15
    v16us au;
#pragma unroll
    for (int i = 0; i < 8; ++i) { au[i] = alo[i]; au[i + 8] = ahi[i]; }
    v16bf a = __builtin_bit_cast(v16bf, au);
#pragma unroll
    for (int j = 0; j < NBLK; ++j) {
      v16us bu = *(const v16us*)(brow[j] + kk);  // K = hi16 + 0..15
      v16bf b = __builtin_bit_cast(v16bf, bu);
      acc[j] = __builtin_amdgcn_wmma_f32_16x16x32_bf16(
          /*neg_a=*/false, a, /*neg_b=*/false, b,
          /*c_mod=*/(short)0, acc[j], /*reuse_a=*/false, /*reuse_b=*/false);
    }
  }

  int rbase = m0 + hi8;          // C/D: vgpr r -> row r + (lane>=16 ? 8 : 0)
#pragma unroll
  for (int j = 0; j < NBLK; ++j) {
    int col = n0 + j * 16 + lrow;
    float bv = bias[col];
#pragma unroll
    for (int r = 0; r < 8; ++r)
      out[(size_t)(rbase + r) * Nout + col] = acc[j][r] + bv;
  }
}

// ---------------------------------------------------------------------------
// h[n, 0..255] = qkvs[n, 768..1023]  (root/skip term)
// ---------------------------------------------------------------------------
__global__ void skip_init(const float* __restrict__ qkvs, float* __restrict__ h, int total) {
  int idx = blockIdx.x * blockDim.x + threadIdx.x;
  if (idx >= total) return;
  int n = idx >> 8, c = idx & 255;
  h[idx] = qkvs[(size_t)n * DCAT + 3 * D1 + c];
}

// ---------------------------------------------------------------------------
// Per-edge attention logits: warp per edge, 8 channels per lane, head = lane/8
// ---------------------------------------------------------------------------
__global__ void edge_logits(const float* __restrict__ qkvs,
                            const int* __restrict__ ei, int E,
                            float* __restrict__ alpha) {
  int gw = (blockIdx.x * blockDim.x + threadIdx.x) >> 5;
  int lane = threadIdx.x & 31;
  if (gw >= E) return;
  int s_ = ei[gw], d_ = ei[E + gw];
  int c0 = lane * 8;
  const float4* qp = (const float4*)(qkvs + (size_t)d_ * DCAT + c0);        // Q of dst
  const float4* kp = (const float4*)(qkvs + (size_t)s_ * DCAT + D1 + c0);   // K of src
  float4 q0 = qp[0], q1 = qp[1], k0 = kp[0], k1 = kp[1];
  float sum = q0.x * k0.x + q0.y * k0.y + q0.z * k0.z + q0.w * k0.w +
              q1.x * k1.x + q1.y * k1.y + q1.z * k1.z + q1.w * k1.w;
  sum += __shfl_down(sum, 4, 8);
  sum += __shfl_down(sum, 2, 8);
  sum += __shfl_down(sum, 1, 8);
  if ((lane & 7) == 0)
    alpha[(size_t)gw * HEADS + (lane >> 3)] = sum * 0.125f;  // 1/sqrt(64)
}

// segment max via order-preserving uint atomicMax
__global__ void seg_max(const float* __restrict__ alpha, const int* __restrict__ ei,
                        int E, unsigned* __restrict__ amax) {
  int idx = blockIdx.x * blockDim.x + threadIdx.x;
  if (idx >= E * HEADS) return;
  int e = idx >> 2, hd = idx & 3;
  int d_ = ei[E + e];
  atomicMax(&amax[(size_t)d_ * HEADS + hd], encf(alpha[idx]));
}

// denom[dst,h] += exp(alpha - amax[dst,h])
__global__ void seg_expsum(const float* __restrict__ alpha, const int* __restrict__ ei,
                           int E, const unsigned* __restrict__ amax,
                           float* __restrict__ denom) {
  int idx = blockIdx.x * blockDim.x + threadIdx.x;
  if (idx >= E * HEADS) return;
  int e = idx >> 2, hd = idx & 3;
  int d_ = ei[E + e];
  float m = decf(amax[(size_t)d_ * HEADS + hd]);
  atomicAdd(&denom[(size_t)d_ * HEADS + hd], __expf(alpha[idx] - m));
}

// h[dst, :] += softmax_weight * V[src, :]   (warp per edge)
__global__ void aggregate(const float* __restrict__ qkvs, const float* __restrict__ alpha,
                          const int* __restrict__ ei, int E,
                          const unsigned* __restrict__ amax,
                          const float* __restrict__ denom,
                          float* __restrict__ h) {
  int gw = (blockIdx.x * blockDim.x + threadIdx.x) >> 5;
  int lane = threadIdx.x & 31;
  if (gw >= E) return;
  int s_ = ei[gw], d_ = ei[E + gw];
  int head = lane >> 3;
  float m = decf(amax[(size_t)d_ * HEADS + head]);
  float w = __expf(alpha[(size_t)gw * HEADS + head] - m) /
            (denom[(size_t)d_ * HEADS + head] + 1e-16f);
  int c0 = lane * 8;
  const float4* vp = (const float4*)(qkvs + (size_t)s_ * DCAT + 2 * D1 + c0);
  float4 v0 = vp[0], v1 = vp[1];
  float* op = h + (size_t)d_ * D1 + c0;
  atomicAdd(op + 0, w * v0.x); atomicAdd(op + 1, w * v0.y);
  atomicAdd(op + 2, w * v0.z); atomicAdd(op + 3, w * v0.w);
  atomicAdd(op + 4, w * v1.x); atomicAdd(op + 5, w * v1.y);
  atomicAdd(op + 6, w * v1.z); atomicAdd(op + 7, w * v1.w);
}

// in-place ELU; optional bf16 copy for the next WMMA GEMM
__global__ void elu_bf(float* __restrict__ h, unsigned short* __restrict__ hbf, int total) {
  int idx = blockIdx.x * blockDim.x + threadIdx.x;
  if (idx >= total) return;
  float f = h[idx];
  float e = (f > 0.f) ? f : expm1f(f);
  h[idx] = e;
  if (hbf) hbf[idx] = f2bf(e);
}

// mean-pool accumulation (atomics into per-graph sums / counts)
__global__ void pool(const float* __restrict__ h, const int* __restrict__ batch,
                     int total, float* __restrict__ sums, float* __restrict__ cnt) {
  int idx = blockIdx.x * blockDim.x + threadIdx.x;
  if (idx >= total) return;
  int n = idx >> 8, c = idx & 255;
  int b = batch[n];
  atomicAdd(&sums[(size_t)b * D1 + c], h[idx]);
  if (c == 0) atomicAdd(&cnt[b], 1.0f);
}

// per-graph classifier + log_softmax (one block of 32 per graph)
__global__ void classify(const float* __restrict__ sums, const float* __restrict__ cnt,
                         const float* __restrict__ Wl, const float* __restrict__ bl,
                         float* __restrict__ out) {
  int g = blockIdx.x, lane = threadIdx.x;
  __shared__ float lg[OUT_CH];
  if (lane < OUT_CH) {
    float cn = fmaxf(cnt[g], 1.0f);
    float acc = bl[lane];
    for (int i = 0; i < D1; ++i)
      acc += (sums[(size_t)g * D1 + i] / cn) * Wl[(size_t)i * OUT_CH + lane];
    lg[lane] = acc;
  }
  __syncthreads();
  if (lane < OUT_CH) {
    float m = -INFINITY;
    for (int j = 0; j < OUT_CH; ++j) m = fmaxf(m, lg[j]);
    float s = 0.f;
    for (int j = 0; j < OUT_CH; ++j) s += __expf(lg[j] - m);
    out[g * OUT_CH + lane] = lg[lane] - m - __logf(s);
  }
}

// ---------------------------------------------------------------------------
// Host launch
// ---------------------------------------------------------------------------
static inline int cdiv(long long a, int b) { return (int)((a + b - 1) / b); }

extern "C" void kernel_launch(void* const* d_in, const int* in_sizes, int n_in,
                              void* d_out, int out_size, void* d_ws, size_t ws_size,
                              hipStream_t stream) {
  const float* x     = (const float*)d_in[0];
  const int*   ei    = (const int*)d_in[1];
  const int*   batch = (const int*)d_in[2];
  const float* Wq1 = (const float*)d_in[3],  *bq1 = (const float*)d_in[4];
  const float* Wk1 = (const float*)d_in[5],  *bk1 = (const float*)d_in[6];
  const float* Wv1 = (const float*)d_in[7],  *bv1 = (const float*)d_in[8];
  const float* Ws1 = (const float*)d_in[9],  *bs1 = (const float*)d_in[10];
  const float* Wq2 = (const float*)d_in[11], *bq2 = (const float*)d_in[12];
  const float* Wk2 = (const float*)d_in[13], *bk2 = (const float*)d_in[14];
  const float* Wv2 = (const float*)d_in[15], *bv2 = (const float*)d_in[16];
  const float* Ws2 = (const float*)d_in[17], *bs2 = (const float*)d_in[18];
  const float* Wl  = (const float*)d_in[19], *bl  = (const float*)d_in[20];
  float* out = (float*)d_out;

  const int N = in_sizes[0] / IN_CH;
  const int E = in_sizes[1] / 2;

  // ---- carve workspace ----
  char* p = (char*)d_ws;
  auto carve = [&](size_t bytes) -> void* {
    void* r = (void*)p;
    p += (bytes + 255) & ~(size_t)255;
    return r;
  };
  unsigned short* Wt1   = (unsigned short*)carve((size_t)DCAT * IN_CH * 2);
  unsigned short* Wt2   = (unsigned short*)carve((size_t)DCAT * D1 * 2);
  float*          bcat1 = (float*)carve(DCAT * 4);
  float*          bcat2 = (float*)carve(DCAT * 4);
  unsigned short* xbf   = (unsigned short*)carve((size_t)N * IN_CH * 2);
  unsigned short* hbf   = (unsigned short*)carve((size_t)N * D1 * 2);
  float*          qkvs  = (float*)carve((size_t)N * DCAT * 4);
  float*          h     = (float*)carve((size_t)N * D1 * 4);
  float*          alpha = (float*)carve((size_t)E * HEADS * 4);
  unsigned*       amax  = (unsigned*)carve((size_t)N * HEADS * 4);
  float*          denom = (float*)carve((size_t)N * HEADS * 4);
  float*          sums  = (float*)carve((size_t)NGRAPH * D1 * 4);
  float*          cnt   = (float*)carve(NGRAPH * 4);

  const int B = 256;
  const int tilesM = (N + 15) / 16;
  const int gemmBlocks = cdiv((long long)tilesM * (DCAT / 64), B / 32);
  const int nd1  = N * D1;       // elementwise over [N, 256]
  const int eh   = E * HEADS;
  const int warpBlocks = cdiv((long long)E * 32, B);

  // ---- weight / input prep (bf16) ----
  prep_wt<<<cdiv((long long)DCAT * IN_CH, B), B, 0, stream>>>(
      Wq1, Wk1, Wv1, Ws1, bq1, bk1, bv1, bs1, Wt1, bcat1, IN_CH);
  prep_wt<<<cdiv((long long)DCAT * D1, B), B, 0, stream>>>(
      Wq2, Wk2, Wv2, Ws2, bq2, bk2, bv2, bs2, Wt2, bcat2, D1);
  f32_to_bf16<<<cdiv((long long)N * IN_CH, B), B, 0, stream>>>(x, xbf, N * IN_CH);

  // ================= layer 1 =================
  gemm_wmma_bf16<<<gemmBlocks, B, 0, stream>>>(xbf, Wt1, bcat1, qkvs, N, IN_CH, DCAT);
  skip_init<<<cdiv(nd1, B), B, 0, stream>>>(qkvs, h, nd1);
  edge_logits<<<warpBlocks, B, 0, stream>>>(qkvs, ei, E, alpha);
  hipMemsetAsync(amax, 0, (size_t)N * HEADS * 4, stream);
  seg_max<<<cdiv(eh, B), B, 0, stream>>>(alpha, ei, E, amax);
  hipMemsetAsync(denom, 0, (size_t)N * HEADS * 4, stream);
  seg_expsum<<<cdiv(eh, B), B, 0, stream>>>(alpha, ei, E, amax, denom);
  aggregate<<<warpBlocks, B, 0, stream>>>(qkvs, alpha, ei, E, amax, denom, h);
  elu_bf<<<cdiv(nd1, B), B, 0, stream>>>(h, hbf, nd1);

  // ================= layer 2 =================
  gemm_wmma_bf16<<<gemmBlocks, B, 0, stream>>>(hbf, Wt2, bcat2, qkvs, N, D1, DCAT);
  skip_init<<<cdiv(nd1, B), B, 0, stream>>>(qkvs, h, nd1);
  edge_logits<<<warpBlocks, B, 0, stream>>>(qkvs, ei, E, alpha);
  hipMemsetAsync(amax, 0, (size_t)N * HEADS * 4, stream);
  seg_max<<<cdiv(eh, B), B, 0, stream>>>(alpha, ei, E, amax);
  hipMemsetAsync(denom, 0, (size_t)N * HEADS * 4, stream);
  seg_expsum<<<cdiv(eh, B), B, 0, stream>>>(alpha, ei, E, amax, denom);
  aggregate<<<warpBlocks, B, 0, stream>>>(qkvs, alpha, ei, E, amax, denom, h);
  elu_bf<<<cdiv(nd1, B), B, 0, stream>>>(h, (unsigned short*)nullptr, nd1);

  // ================= pool + classify =================
  hipMemsetAsync(sums, 0, (size_t)NGRAPH * D1 * 4, stream);
  hipMemsetAsync(cnt, 0, NGRAPH * 4, stream);
  pool<<<cdiv(nd1, B), B, 0, stream>>>(h, batch, nd1, sums, cnt);
  classify<<<NGRAPH, 32, 0, stream>>>(sums, cnt, Wl, bl, out);
}
